// SVDQuantLinear_67439576481968
// MI455X (gfx1250) — 16-bit and 8-bit matrix operand order repaired and checked by reading, compiled, not run on hardware
//
#include <hip/hip_runtime.h>
#include <hip/hip_bf16.h>
#include <stdint.h>

#define MDIM 4096
#define NDIM 4096
#define KDIM 4096
#define GRP  64
#define NGRP (KDIM / GRP)
#define RANK 32

typedef int      v4i  __attribute__((ext_vector_type(4)));
typedef int      v8i  __attribute__((ext_vector_type(8)));
typedef float    v8f  __attribute__((ext_vector_type(8)));
typedef _Float16 v16h __attribute__((ext_vector_type(16)));

#define AS1 __attribute__((address_space(1)))
#define AS3 __attribute__((address_space(3)))

__global__ void __launch_bounds__(256)
quant_pack_a(const float* __restrict__ x,
             const float* __restrict__ smooth,
             uint8_t* __restrict__ Apack,
             float* __restrict__ ascT) {
  const int m   = blockIdx.x;
  const int tid = threadIdx.x;
  const int k0  = tid * 16;

  float vals[16];
  float amax = 0.0f;
  const float4* xp = (const float4*)(x + (size_t)m * KDIM + k0);
  const float4* sp = (const float4*)(smooth + k0);
#pragma unroll
  for (int t = 0; t < 4; ++t) {
    float4 xv = xp[t];
    float4 sv = sp[t];
    float a = xv.x / sv.x, b = xv.y / sv.y, c = xv.z / sv.z, d = xv.w / sv.w;
    vals[t * 4 + 0] = a; vals[t * 4 + 1] = b;
    vals[t * 4 + 2] = c; vals[t * 4 + 3] = d;
    amax = fmaxf(amax, fmaxf(fmaxf(fabsf(a), fabsf(b)), fmaxf(fabsf(c), fabsf(d))));
  }
  amax = fmaxf(amax, __shfl_xor(amax, 1, 32));
  amax = fmaxf(amax, __shfl_xor(amax, 2, 32));
  const float ascale = fmaxf(amax * (1.0f / 7.0f), 1e-8f);

  const int g = tid >> 2;
  if ((tid & 3) == 0) ascT[(size_t)g * MDIM + m] = ascale;

  const float inv = 1.0f / ascale;
  unsigned long long lo = 0ull, hi = 0ull;
#pragma unroll
  for (int i = 0; i < 8; ++i) {
    int q = (int)rintf(vals[i] * inv);
    q = q < -8 ? -8 : (q > 7 ? 7 : q);
    lo |= (unsigned long long)(uint8_t)(int8_t)q << (8 * i);
  }
#pragma unroll
  for (int i = 0; i < 8; ++i) {
    int q = (int)rintf(vals[8 + i] * inv);
    q = q < -8 ? -8 : (q > 7 ? 7 : q);
    hi |= (unsigned long long)(uint8_t)(int8_t)q << (8 * i);
  }

  const int q4 = tid & 3;
  uint8_t* tb = Apack + (((size_t)(m >> 4) * NGRP + g) << 10);
  *(unsigned long long*)(tb + ((m & 15) + 16 * (q4 & 1)) * 32 + 16 * (q4 >> 1))     = lo;
  *(unsigned long long*)(tb + ((m & 15) + 16 * (q4 & 1)) * 32 + 16 * (q4 >> 1) + 8) = hi;
}

__global__ void __launch_bounds__(256)
pack_b(const int* __restrict__ qw, uint8_t* __restrict__ Bpack) {
  const int idx = blockIdx.x * 256 + threadIdx.x;
  const int n   = idx & (NDIM - 1);
  const int b16 = idx >> 12;
  const int kb  = b16 * 16;
  const int g   = kb >> 6;
  const int b   = (kb >> 4) & 3;

  uint32_t w[4] = {0u, 0u, 0u, 0u};
#pragma unroll
  for (int i = 0; i < 16; ++i) {
    int v = qw[(size_t)(kb + i) * NDIM + n];
    w[i >> 2] |= (uint32_t)(uint8_t)(int8_t)v << (8 * (i & 3));
  }
  uint8_t* tb = Bpack + (((size_t)(n >> 4) * NGRP + g) << 10);
  const int lanep = (n & 15) + 16 * (b & 1);
  *(uint4*)(tb + lanep * 32 + 16 * (b >> 1)) = make_uint4(w[0], w[1], w[2], w[3]);
}

__global__ void __launch_bounds__(256)
lora_xt_kernel(const float* __restrict__ x,
               const float* __restrict__ ld,
               float* __restrict__ xt) {
  const int lane = threadIdx.x & 31;
  const int m    = blockIdx.x * 8 + (threadIdx.x >> 5);
  float acc = 0.0f;
  for (int kb = 0; kb < KDIM; kb += 32) {
    float xv = x[(size_t)m * KDIM + kb + lane];
#pragma unroll
    for (int j = 0; j < 32; ++j) {
      acc = fmaf(__shfl(xv, j, 32), ld[(size_t)(kb + j) * RANK + lane], acc);
    }
  }
  xt[(size_t)m * RANK + lane] = acc;
}

__global__ void __launch_bounds__(256)
gemm_int4_lora(const uint8_t* __restrict__ Apack,
               const uint8_t* __restrict__ Bpack,
               const float* __restrict__ ascT,
               const float* __restrict__ wsc,
               const float* __restrict__ xt,
               const float* __restrict__ up,
               const float* __restrict__ bias,
               float* __restrict__ out) {
  __shared__ __attribute__((aligned(16))) uint8_t lds[3][16384];

  const int tid  = threadIdx.x;
  const int lane = tid & 31;
  const int wave = tid >> 5;
  const int half = lane >> 4;
  const int l16  = lane & 15;
  const int bn   = blockIdx.x & 31;
  const int bm   = blockIdx.x >> 5;
  const int m0   = bm * 128 + (wave >> 1) * 32;
  const int n0   = bn * 128 + (wave & 1) * 64;
  const int mtb  = bm * 8;
  const int ntb  = bn * 8;
  const int ta0  = (wave >> 1) * 2;
  const int tb0  = (wave & 1) * 4;

  const int o0 = tid * 16;
  const int o1 = o0 + 4096;
  const int tt0 = o0 >> 10, w0o = o0 & 1023;
  const int tt1 = o1 >> 10, w1o = o1 & 1023;
  const uint8_t* gA0 = Apack + (((size_t)(mtb + tt0) * NGRP) << 10) + w0o;
  const uint8_t* gA1 = Apack + (((size_t)(mtb + tt1) * NGRP) << 10) + w1o;
  const uint8_t* gB0 = Bpack + (((size_t)(ntb + tt0) * NGRP) << 10) + w0o;
  const uint8_t* gB1 = Bpack + (((size_t)(ntb + tt1) * NGRP) << 10) + w1o;

#define STAGE_ASYNC(buf, g)                                                     \
  do {                                                                          \
    uint8_t* lb = &lds[(buf)][0];                                               \
    __builtin_amdgcn_global_load_async_to_lds_b128(                             \
        (AS1 v4i*)(gA0 + ((size_t)(g) << 10)),                                  \
        (AS3 v4i*)(lb + o0), 0, 0);                                             \
    __builtin_amdgcn_global_load_async_to_lds_b128(                             \
        (AS1 v4i*)(gA1 + ((size_t)(g) << 10)),                                  \
        (AS3 v4i*)(lb + o1), 0, 0);                                             \
    __builtin_amdgcn_global_load_async_to_lds_b128(                             \
        (AS1 v4i*)(gB0 + ((size_t)(g) << 10)),                                  \
        (AS3 v4i*)(lb + 8192 + o0), 0, 0);                                      \
    __builtin_amdgcn_global_load_async_to_lds_b128(                             \
        (AS1 v4i*)(gB1 + ((size_t)(g) << 10)),                                  \
        (AS3 v4i*)(lb + 8192 + o1), 0, 0);                                      \
  } while (0)

  v8f acc[2][4];
  const v8f vzero = {0.f, 0.f, 0.f, 0.f, 0.f, 0.f, 0.f, 0.f};
#pragma unroll
  for (int ii = 0; ii < 2; ++ii)
#pragma unroll
    for (int jj = 0; jj < 4; ++jj) acc[ii][jj] = vzero;

  const v8i zero = {0, 0, 0, 0, 0, 0, 0, 0};

  STAGE_ASYNC(0, 0);
  STAGE_ASYNC(1, 1);

  int cb = 0;
  int sb = 2;

  for (int g = 0; g < NGRP; ++g) {
    if (g + 1 < NGRP) {
      asm volatile("s_wait_asynccnt 0x4" ::: "memory");
    } else {
      asm volatile("s_wait_asynccnt 0x0" ::: "memory");
    }
    __syncthreads();
    if (g + 2 < NGRP) STAGE_ASYNC(sb, g + 2);

    const uint8_t* cur = &lds[cb][0];

    v8i a[2], b[4];
#pragma unroll
    for (int ii = 0; ii < 2; ++ii)
      a[ii] = *(const v8i*)(cur + (ta0 + ii) * 1024 + lane * 32);
#pragma unroll
    for (int jj = 0; jj < 4; ++jj)
      b[jj] = *(const v8i*)(cur + 8192 + (tb0 + jj) * 1024 + lane * 32);

    float ascv[2][8];
#pragma unroll
    for (int ii = 0; ii < 2; ++ii) {
      const float* ap = ascT + (size_t)g * MDIM + m0 + ii * 16 + 8 * half;
      float4 p0 = *(const float4*)ap;
      float4 p1 = *(const float4*)(ap + 4);
      ascv[ii][0] = p0.x; ascv[ii][1] = p0.y; ascv[ii][2] = p0.z; ascv[ii][3] = p0.w;
      ascv[ii][4] = p1.x; ascv[ii][5] = p1.y; ascv[ii][6] = p1.z; ascv[ii][7] = p1.w;
    }
    float w[4];
#pragma unroll
    for (int jj = 0; jj < 4; ++jj)
      w[jj] = wsc[(size_t)g * NDIM + n0 + jj * 16 + l16];

    v8i sres[2][4];
#pragma unroll
    for (int ii = 0; ii < 2; ++ii)
#pragma unroll
      for (int jj = 0; jj < 4; ++jj)
        sres[ii][jj] = __builtin_amdgcn_wmma_i32_16x16x64_iu8(
            true, a[ii], true, b[jj], zero, false, false);
#pragma unroll
    for (int ii = 0; ii < 2; ++ii)
#pragma unroll
      for (int jj = 0; jj < 4; ++jj)
#pragma unroll
        for (int v = 0; v < 8; ++v)
          acc[ii][jj][v] = fmaf((float)sres[ii][jj][v] * ascv[ii][v], w[jj],
                                acc[ii][jj][v]);

    cb = (cb == 2) ? 0 : cb + 1;
    sb = (sb == 2) ? 0 : sb + 1;
  }

  v16h Ah[2], Al[2], Bh[4], Bl[4];
#pragma unroll
  for (int ii = 0; ii < 2; ++ii) {
    const float* xr = xt + (size_t)(m0 + ii * 16 + l16) * RANK;
    float f[16];
    float4 q0 = *(const float4*)(xr + 8 * half);
    float4 q1 = *(const float4*)(xr + 8 * half + 4);
    float4 q2 = *(const float4*)(xr + 16 + 8 * half);
    float4 q3 = *(const float4*)(xr + 16 + 8 * half + 4);
    f[0] = q0.x; f[1] = q0.y; f[2]  = q0.z; f[3]  = q0.w;
    f[4] = q1.x; f[5] = q1.y; f[6]  = q1.z; f[7]  = q1.w;
    f[8] = q2.x; f[9] = q2.y; f[10] = q2.z; f[11] = q2.w;
    f[12] = q3.x; f[13] = q3.y; f[14] = q3.z; f[15] = q3.w;
#pragma unroll
    for (int e = 0; e < 16; ++e) {
      _Float16 h = (_Float16)f[e];
      Ah[ii][e] = h;
      Al[ii][e] = (_Float16)(f[e] - (float)h);
    }
  }
#pragma unroll
  for (int jj = 0; jj < 4; ++jj) {
    const float* uc = up + n0 + jj * 16 + l16;
#pragma unroll
    for (int e = 0; e < 16; ++e) {
      float u = uc[(size_t)(e < 8 ? 8 * half + e : 8 + 8 * half + e) * NDIM];
      _Float16 h = (_Float16)u;
      Bh[jj][e] = h;
      Bl[jj][e] = (_Float16)(u - (float)h);
    }
  }
#pragma unroll
  for (int ii = 0; ii < 2; ++ii)
#pragma unroll
    for (int jj = 0; jj < 4; ++jj) {
      v8f c = acc[ii][jj];
      c = __builtin_amdgcn_wmma_f32_16x16x32_f16(false, Ah[ii], false, Bh[jj],
                                                 (short)0, c, false, false);
      c = __builtin_amdgcn_wmma_f32_16x16x32_f16(false, Al[ii], false, Bh[jj],
                                                 (short)0, c, false, false);
      c = __builtin_amdgcn_wmma_f32_16x16x32_f16(false, Ah[ii], false, Bl[jj],
                                                 (short)0, c, false, false);
      acc[ii][jj] = c;
    }

  float bb[4];
#pragma unroll
  for (int jj = 0; jj < 4; ++jj) bb[jj] = bias[n0 + jj * 16 + l16];
#pragma unroll
  for (int ii = 0; ii < 2; ++ii)
#pragma unroll
    for (int v = 0; v < 8; ++v) {
      const int row = m0 + ii * 16 + v + 8 * half;
#pragma unroll
      for (int jj = 0; jj < 4; ++jj)
        out[(size_t)row * NDIM + n0 + jj * 16 + l16] = acc[ii][jj][v] + bb[jj];
    }
#undef STAGE_ASYNC
}

extern "C" void kernel_launch(void* const* d_in, const int* in_sizes, int n_in,
                              void* d_out, int out_size, void* d_ws, size_t ws_size,
                              hipStream_t stream) {
  (void)in_sizes; (void)n_in; (void)out_size; (void)ws_size;
  const float* x    = (const float*)d_in[0];
  const int*   qw   = (const int*)d_in[1];
  const float* wsc  = (const float*)d_in[2];
  const float* ld   = (const float*)d_in[3];
  const float* up   = (const float*)d_in[4];
  const float* sm   = (const float*)d_in[5];
  const float* bias = (const float*)d_in[6];
  float* out = (float*)d_out;

  uint8_t* ws    = (uint8_t*)d_ws;
  uint8_t* Apack = ws;
  uint8_t* Bpack = ws + ((size_t)16 << 20);
  float*   ascT  = (float*)(ws + ((size_t)32 << 20));
  float*   xt    = (float*)(ws + ((size_t)33 << 20));

  quant_pack_a<<<MDIM, 256, 0, stream>>>(x, sm, Apack, ascT);
  pack_b<<<(KDIM / 16) * NDIM / 256, 256, 0, stream>>>(qw, Bpack);
  lora_xt_kernel<<<MDIM / 8, 256, 0, stream>>>(x, ld, xt);
  gemm_int4_lora<<<(MDIM / 128) * (NDIM / 128), 256, 0, stream>>>(
      Apack, Bpack, ascT, wsc, xt, up, bias, out);
}
